// BailingMoeV2_5LinearAttention_67929202753679
// MI455X (gfx1250) — matrix operand order repaired and checked by reading, not compiled, not run on hardware
//
#include <hip/hip_runtime.h>
#include <cstdint>
#include <cstddef>

// ---------------------------------------------------------------------------
// Types for CDNA5 WMMA (wave32, v_wmma_f32_16x16x32_bf16)
// ---------------------------------------------------------------------------
typedef __bf16 bf16_t;
typedef __attribute__((ext_vector_type(16))) __bf16        v16bf;
typedef __attribute__((ext_vector_type(8)))  float         v8f;
typedef __attribute__((ext_vector_type(4)))  unsigned int  u32x4;
typedef __attribute__((ext_vector_type(4)))  unsigned int  v4u;
typedef __attribute__((ext_vector_type(8)))  int           v8i;
typedef __attribute__((ext_vector_type(4)))  int           v4i;

#define NHEAD   16
#define HDIM    128
#define HIDDEN  2048
#define SEQLEN  2048
#define BATCH   2
#define CHUNKL  64
#define NCHUNK  32
#define NHD     2048      /* NHEAD*HDIM */
#define QKVN    6144      /* 3*NHD      */

union FragB16 { v16bf v; u32x4 q[2]; };
union Half8   { u32x4 v; bf16_t h[8]; };

__device__ __forceinline__ v8f wmma_bf16(v16bf a, v16bf b, v8f c) {
    return __builtin_amdgcn_wmma_f32_16x16x32_bf16(false, a, false, b,
                                                   (short)0, c, false, false);
}

// ---------------------------------------------------------------------------
// Tensor Data Mover: async 2D tile load (row-major, bf16 elements) -> LDS.
// D# built per CDNA5 ISA ch.8 (group0/group1 bitfields), groups 2/3 unused.
// This toolchain exposes the 6-arg builtin (g0, g1, g2, g3, g4, cpol).
// Issue from ONE wave; that wave must s_wait_tensorcnt(0) before the barrier.
// ---------------------------------------------------------------------------
#if __has_builtin(__builtin_amdgcn_tensor_load_to_lds) && \
    __has_builtin(__builtin_amdgcn_s_wait_tensorcnt)
#define HAVE_TDM 1
#else
#define HAVE_TDM 0
#endif

#if HAVE_TDM
__device__ __forceinline__ void tdm_load_2d_bf16(void* lds_ptr, const void* gptr,
                                                 unsigned rows, unsigned cols,
                                                 unsigned ld_elems) {
    const unsigned lds_off = (unsigned)(uintptr_t)lds_ptr;   // LDS aperture: low 32 bits = LDS byte offset
    const unsigned long long ga = (unsigned long long)(uintptr_t)gptr;

    v4u g0;
    g0[0] = 1u;                                   // count=1, is_restore=0, no gather
    g0[1] = lds_off;                              // lds_addr (bytes)
    g0[2] = (unsigned)(ga & 0xffffffffu);         // global_addr[31:0]
    g0[3] = (unsigned)((ga >> 32) & 0x01ffffffu)  // global_addr[56:32]
          | (2u << 30);                           // type = 2 ("image")

    v8i g1;
    g1[0] = (int)(1u << 16);                      // workgroup_mask=0, data_size=1 (2B)
    g1[1] = (int)((cols & 0xffffu) << 16);        // tensor_dim0[15:0] (atomic_barrier_addr=0)
    g1[2] = (int)(((cols >> 16) & 0xffffu) | ((rows & 0xffffu) << 16)); // td0[31:16], td1[15:0]
    g1[3] = (int)(((rows >> 16) & 0xffffu) | ((cols & 0xffffu) << 16)); // td1[31:16], tile_dim0
    g1[4] = (int)(rows & 0xffffu);                // tile_dim1 (tile_dim2=0)
    g1[5] = (int)ld_elems;                        // tensor_dim0_stride[31:0]
    g1[6] = 0;                                    // stride0[47:32]=0, stride1[15:0]=0
    g1[7] = 0;                                    // stride1[47:16]=0

    const v4i z4 = (v4i)0;
    const v8i z8 = (v8i)0;
    __builtin_amdgcn_tensor_load_to_lds(g0, g1, z4, z4, z8, 0);
}
#define TDM_WAIT() __builtin_amdgcn_s_wait_tensorcnt(0)
#endif

// Fallback / generic vectorized 2D tile copy: 16B chunks, 256 threads.
__device__ __forceinline__ void copy_tile_2d(bf16_t* dst, const bf16_t* src,
                                             int rows, int cols, int ld,
                                             int tid) {
    const int chunks = rows * (cols >> 3);
    const int cpr = cols >> 3;                    // 16B chunks per row
    for (int idx = tid; idx < chunks; idx += 256) {
        const int r = idx / cpr, c = idx % cpr;
        *(u32x4*)(dst + r * cols + c * 8) =
            *(const u32x4*)(src + (size_t)r * ld + c * 8);
    }
}

// A-fragment (16x32 bf16) from row-major tile (element (m,k)), ISA 7.12.2.
__device__ __forceinline__ v16bf load_a_frag(const bf16_t* tile, int m0, int k0,
                                             int stride, int lane) {
    FragB16 f;
    const int m   = m0 + (lane & 15);
    const int seg = (lane >> 4) << 3;             // 0 or 8 halves
    const bf16_t* p = tile + m * stride + k0 + seg;
    f.q[0] = *(const u32x4*)(p);
    f.q[1] = *(const u32x4*)(p + 16);
    return f.v;
}

// B-fragment (32x16 bf16) from a B^T tile stored [n][stride] (element=B[k][n]).
__device__ __forceinline__ v16bf load_b_frag(const bf16_t* tileT, int n0, int k0,
                                             int stride, int lane) {
    FragB16 f;
    const int n   = n0 + (lane & 15);
    const int seg = (lane >> 4) << 3;             // 0 or 8 halves
    const bf16_t* p = tileT + n * stride + k0 + seg;
    f.q[0] = *(const u32x4*)(p);
    f.q[1] = *(const u32x4*)(p + 16);
    return f.v;
}

// A-fragment with per-row scaling (q * q_dec[i]).
__device__ __forceinline__ v16bf load_a_frag_scaled(const bf16_t* tile, int m0,
                                                    int k0, int stride, int lane,
                                                    float scale) {
    const int m   = m0 + (lane & 15);
    const int seg = (lane >> 4) << 3;
    const bf16_t* p = tile + m * stride + k0 + seg;
    v16bf v;
#pragma unroll
    for (int e = 0; e < 8; ++e) {
        v[e]     = (bf16_t)((float)p[e]      * scale);
        v[e + 8] = (bf16_t)((float)p[16 + e] * scale);
    }
    return v;
}

// B-fragment built on the fly from an fp32 B^T tile (recurrent state).
__device__ __forceinline__ v16bf load_b_frag_f32(const float* tileT, int n0,
                                                 int k0, int stride, int lane) {
    const int n = n0 + (lane & 15);
    const float* p = tileT + n * stride + k0 + ((lane >> 4) << 3);
    v16bf v;
#pragma unroll
    for (int e = 0; e < 16; ++e) v[e] = (bf16_t)p[(e & 7) + ((e >> 3) << 4)];
    return v;
}

// ---------------------------------------------------------------------------
// fp32 -> bf16 conversion
// ---------------------------------------------------------------------------
__global__ __launch_bounds__(256) void cvt_f32_bf16(const float* __restrict__ in,
                                                    bf16_t* __restrict__ out,
                                                    size_t n) {
    size_t i = (size_t)blockIdx.x * 256u + threadIdx.x;
    const size_t stride = (size_t)gridDim.x * 256u;
    for (; i < n; i += stride) out[i] = (bf16_t)in[i];
}

// ---------------------------------------------------------------------------
// bf16 GEMM: C[M,N] = A[M,K] @ B[K,N], fp32 accumulation.
// Block tile 128x128, 256 threads (8 waves), K-step 64, double-buffered LDS.
// A tile staged by the Tensor Data Mover (wave 0), overlapped with compute;
// B tile: wide global loads issued before compute, transpose-scatter stores
// after compute. One of Cf/Cb is non-null.
// ---------------------------------------------------------------------------
__global__ __launch_bounds__(256) void gemm_bf16(const bf16_t* __restrict__ A,
                                                 const bf16_t* __restrict__ Bw,
                                                 float* __restrict__ Cf,
                                                 bf16_t* __restrict__ Cb,
                                                 int M, int N, int K) {
    __shared__ bf16_t As[2][128 * 64];   // row-major (m,k)
    __shared__ bf16_t Bt[2][128 * 64];   // transposed: (n,k)

    const int tid  = threadIdx.x;
    const int lane = tid & 31;
    const int wave = tid >> 5;
    const int bm = blockIdx.y * 128;
    const int bn = blockIdx.x * 128;
    const int wm = (wave & 3) * 32;   // 4 waves along M
    const int wn = (wave >> 2) * 64;  // 2 waves along N

    v8f acc[2][4];
#pragma unroll
    for (int i = 0; i < 2; ++i)
#pragma unroll
        for (int j = 0; j < 4; ++j)
            acc[i][j] = (v8f){0.f, 0.f, 0.f, 0.f, 0.f, 0.f, 0.f, 0.f};

    const int NT = K >> 6;  // K / 64 tiles

    // ---- Prologue: stage tile 0 into buffer 0 ----
#if HAVE_TDM
    if (wave == 0)
        tdm_load_2d_bf16(As[0], &A[(size_t)bm * K], 128, 64, (unsigned)K);
#else
    copy_tile_2d(As[0], &A[(size_t)bm * K], 128, 64, K, tid);
#endif
    {
        Half8 breg[4];
        int bk[4], bn0[4];
#pragma unroll
        for (int r = 0; r < 4; ++r) {
            const int idx = r * 256 + tid;
            bk[r]  = idx >> 4;
            bn0[r] = (idx & 15) * 8;
            breg[r].v = *(const u32x4*)&Bw[(size_t)bk[r] * N + bn + bn0[r]];
        }
#pragma unroll
        for (int r = 0; r < 4; ++r)
#pragma unroll
            for (int e = 0; e < 8; ++e)
                Bt[0][(bn0[r] + e) * 64 + bk[r]] = breg[r].h[e];
    }
#if HAVE_TDM
    if (wave == 0) TDM_WAIT();
#endif
    __syncthreads();

    for (int t = 0; t < NT; ++t) {
        const int cur = t & 1, nxt = cur ^ 1;
        const bool has_next = (t + 1) < NT;

        // ---- Issue next tile's global traffic before computing ----
        Half8 breg[4];
        int bk[4], bn0[4];
        if (has_next) {
            const int kn = (t + 1) << 6;
#pragma unroll
            for (int r = 0; r < 4; ++r) {
                const int idx = r * 256 + tid;
                bk[r]  = idx >> 4;
                bn0[r] = (idx & 15) * 8;
                breg[r].v = *(const u32x4*)&Bw[(size_t)(kn + bk[r]) * N + bn + bn0[r]];
            }
#if HAVE_TDM
            if (wave == 0)
                tdm_load_2d_bf16(As[nxt], &A[(size_t)bm * K + kn], 128, 64,
                                 (unsigned)K);
#else
            copy_tile_2d(As[nxt], &A[(size_t)bm * K + kn], 128, 64, K, tid);
#endif
        }

        // ---- Compute current tile: all fragments hoisted per k-step ----
#pragma unroll
        for (int k2 = 0; k2 < 64; k2 += 32) {
            const v16bf a0 = load_a_frag(As[cur], wm,      k2, 64, lane);
            const v16bf a1 = load_a_frag(As[cur], wm + 16, k2, 64, lane);
            v16bf bfr[4];
#pragma unroll
            for (int j = 0; j < 4; ++j)
                bfr[j] = load_b_frag(Bt[cur], wn + j * 16, k2, 64, lane);
#pragma unroll
            for (int j = 0; j < 4; ++j) {
                acc[0][j] = wmma_bf16(a0, bfr[j], acc[0][j]);
                acc[1][j] = wmma_bf16(a1, bfr[j], acc[1][j]);
            }
        }

        // ---- Land next B tile into LDS, finish TDM, flip ----
        if (has_next) {
#pragma unroll
            for (int r = 0; r < 4; ++r)
#pragma unroll
                for (int e = 0; e < 8; ++e)
                    Bt[nxt][(bn0[r] + e) * 64 + bk[r]] = breg[r].h[e];
#if HAVE_TDM
            if (wave == 0) TDM_WAIT();
#endif
        }
        __syncthreads();
    }

#pragma unroll
    for (int i = 0; i < 2; ++i)
#pragma unroll
        for (int j = 0; j < 4; ++j)
#pragma unroll
            for (int r = 0; r < 8; ++r) {
                const int m = bm + wm + i * 16 + r + ((lane >> 4) << 3);
                const int n = bn + wn + j * 16 + (lane & 15);
                const float v = acc[i][j][r];
                if (Cb) Cb[(size_t)m * N + n] = (bf16_t)v;
                else    Cf[(size_t)m * N + n] = v;
            }
}

// ---------------------------------------------------------------------------
// Per-(b,s,h) RMSNorm + partial RoPE for q and k. 128 threads per block.
// ---------------------------------------------------------------------------
__global__ __launch_bounds__(128) void qk_norm_rope(const bf16_t* __restrict__ qkv,
                                                    const int* __restrict__ pos,
                                                    const float* __restrict__ qw,
                                                    const float* __restrict__ kw,
                                                    bf16_t* __restrict__ qr,
                                                    bf16_t* __restrict__ kr) {
    const int idx = blockIdx.x;
    const int h = idx & 15;
    const int s = (idx >> 4) & (SEQLEN - 1);
    const int b = idx >> 15;
    const int d = threadIdx.x;
    const size_t row = (size_t)b * SEQLEN + s;

    const float qv = (float)qkv[row * QKVN +       h * HDIM + d];
    const float kv = (float)qkv[row * QKVN + NHD + h * HDIM + d];

    __shared__ float redq[4], redk[4];
    __shared__ float shq[128], shk[128];

    float sq = qv * qv, sk = kv * kv;
#pragma unroll
    for (int off = 16; off; off >>= 1) {
        sq += __shfl_xor(sq, off);
        sk += __shfl_xor(sk, off);
    }
    if ((d & 31) == 0) { redq[d >> 5] = sq; redk[d >> 5] = sk; }
    __syncthreads();
    const float sumq = redq[0] + redq[1] + redq[2] + redq[3];
    const float sumk = redk[0] + redk[1] + redk[2] + redk[3];
    const float rq = rsqrtf(sumq * (1.0f / 128.0f) + 1e-6f);
    const float rk = rsqrtf(sumk * (1.0f / 128.0f) + 1e-6f);

    shq[d] = qv * rq * qw[d];
    shk[d] = kv * rk * kw[d];
    __syncthreads();

    const float fp = (float)pos[s];
    float oq, ok;
    if (d < 64) {
        const int j = d & 31;
        const float ang = fp * powf(10000.0f, -(float)j / 32.0f);
        const float cs = cosf(ang), sn = sinf(ang);
        if (d < 32) { oq = shq[d] * cs - shq[d + 32] * sn;
                      ok = shk[d] * cs - shk[d + 32] * sn; }
        else        { oq = shq[d] * cs + shq[d - 32] * sn;
                      ok = shk[d] * cs + shk[d - 32] * sn; }
    } else { oq = shq[d]; ok = shk[d]; }

    qr[row * NHD + h * HDIM + d] = (bf16_t)oq;
    kr[row * NHD + h * HDIM + d] = (bf16_t)ok;
}

// ---------------------------------------------------------------------------
// Chunked linear attention. grid = B*NH blocks, 256 threads (8 waves).
// stateT[e][d] (fp32) in LDS; q/k chunk tiles staged by TDM (waves 0/1).
// ---------------------------------------------------------------------------
__global__ __launch_bounds__(256) void attn_kernel(const bf16_t* __restrict__ qr,
                                                   const bf16_t* __restrict__ kr,
                                                   const bf16_t* __restrict__ qkv,
                                                   const float* __restrict__ rec,
                                                   float* __restrict__ o) {
    extern __shared__ char smem[];
    float*  stateT = (float*)smem;                            // [128][128] f32
    bf16_t* q_rm = (bf16_t*)(smem + 65536);                   // [64][128]
    bf16_t* k_rm = q_rm + 64 * 128;                           // [64][128]
    bf16_t* kdT  = k_rm + 64 * 128;                           // [128][64]
    bf16_t* vT   = kdT + 128 * 64;                            // [128][64]
    bf16_t* s_bf = vT  + 128 * 64;                            // [64][64]
    float*  qdec = (float*)(s_bf + 64 * 64);                  // [64]
    float*  kdec = qdec + 64;                                 // [64]

    const int tid = threadIdx.x, lane = tid & 31, wave = tid >> 5;
    const int b = blockIdx.x >> 4, h = blockIdx.x & 15;

    const float slope = -powf(0.70710678118654752f, (float)(h + 1)) * (1.0f + 1e-5f);
    const float lam   = expf(slope * (float)CHUNKL);

    const float* rs = rec + ((size_t)(b * NHEAD + h)) * HDIM * HDIM;
    for (int i = tid; i < 128 * 128; i += 256) {
        const int e = i & 127, d = i >> 7;
        stateT[e * 128 + d] = rs[d * 128 + e];
    }
    if (tid < 64) {
        qdec[tid] = expf(slope * (float)(tid + 1));
        kdec[tid] = expf(slope * (float)(CHUNKL - 1 - tid));
    }
    __syncthreads();

    for (int c = 0; c < NCHUNK; ++c) {
        const size_t rowbase = (size_t)b * SEQLEN + (size_t)c * CHUNKL;
        const bf16_t* qg = qr  + rowbase * NHD + h * HDIM;
        const bf16_t* kg = kr  + rowbase * NHD + h * HDIM;
        const bf16_t* vg = qkv + rowbase * QKVN + 2 * NHD + h * HDIM;

        // ---- Stage q, k (row-major, TDM) and v (transposed, manual) ----
#if HAVE_TDM
        if (wave == 0) tdm_load_2d_bf16(q_rm, qg, 64, 128, NHD);
        if (wave == 1) tdm_load_2d_bf16(k_rm, kg, 64, 128, NHD);
#else
        copy_tile_2d(q_rm, qg, 64, 128, NHD, tid);
        copy_tile_2d(k_rm, kg, 64, 128, NHD, tid);
#endif
        {
            Half8 vreg[4];
            int vr[4], vd0[4];
#pragma unroll
            for (int t = 0; t < 4; ++t) {
                const int idx = t * 256 + tid;        // 1024 chunks of 8
                vr[t]  = idx >> 4;                    // row j 0..63
                vd0[t] = (idx & 15) * 8;              // d chunk
                vreg[t].v = *(const u32x4*)&vg[(size_t)vr[t] * QKVN + vd0[t]];
            }
#pragma unroll
            for (int t = 0; t < 4; ++t)
#pragma unroll
                for (int e = 0; e < 8; ++e)
                    vT[(vd0[t] + e) * 64 + vr[t]] = vreg[t].h[e];
        }
#if HAVE_TDM
        if (wave < 2) TDM_WAIT();
#endif
        __syncthreads();

        // kdT[d][j] = k[j][d] * k_dec[j] (wide LDS reads, scatter stores)
        {
            Half8 kreg[4];
            int kj[4], kd0[4];
#pragma unroll
            for (int t = 0; t < 4; ++t) {
                const int idx = t * 256 + tid;
                kj[t]  = idx >> 4;
                kd0[t] = (idx & 15) * 8;
                kreg[t].v = *(const u32x4*)&k_rm[kj[t] * 128 + kd0[t]];
            }
#pragma unroll
            for (int t = 0; t < 4; ++t) {
                const float sc = kdec[kj[t]];
#pragma unroll
                for (int e = 0; e < 8; ++e)
                    kdT[(kd0[t] + e) * 64 + kj[t]] =
                        (bf16_t)((float)kreg[t].h[e] * sc);
            }
        }
        __syncthreads();

        // ---- Phase 1: scores = q @ k^T with decay mask -> s_bf ----
        {
            const int mi = wave >> 1, nbase = (wave & 1) * 2;
#pragma unroll
            for (int t = 0; t < 2; ++t) {
                const int ni = nbase + t;
                v8f cacc = (v8f){0.f, 0.f, 0.f, 0.f, 0.f, 0.f, 0.f, 0.f};
#pragma unroll
                for (int kk = 0; kk < 128; kk += 32) {
                    const v16bf a  = load_a_frag(q_rm, mi * 16, kk, 128, lane);
                    const v16bf bb = load_b_frag(k_rm, ni * 16, kk, 128, lane);
                    cacc = wmma_bf16(a, bb, cacc);
                }
#pragma unroll
                for (int r = 0; r < 8; ++r) {
                    const int i_ = mi * 16 + r + ((lane >> 4) << 3);
                    const int j_ = ni * 16 + (lane & 15);
                    const float v = (i_ >= j_)
                        ? cacc[r] * expf(slope * (float)(i_ - j_)) : 0.0f;
                    s_bf[i_ * 64 + j_] = (bf16_t)v;
                }
            }
        }
        __syncthreads();

        // ---- Phase 2: o = scores @ v + (q*q_dec) @ state ----
        {
            const int mi = wave & 3;
            const int nb = (wave >> 2) * 4;
            v8f oc[4];
#pragma unroll
            for (int j = 0; j < 4; ++j)
                oc[j] = (v8f){0.f, 0.f, 0.f, 0.f, 0.f, 0.f, 0.f, 0.f};
#pragma unroll
            for (int kk = 0; kk < 64; kk += 32) {
                const v16bf a = load_a_frag(s_bf, mi * 16, kk, 64, lane);
                v16bf bfr[4];
#pragma unroll
                for (int j = 0; j < 4; ++j)
                    bfr[j] = load_b_frag(vT, (nb + j) * 16, kk, 64, lane);
#pragma unroll
                for (int j = 0; j < 4; ++j)
                    oc[j] = wmma_bf16(a, bfr[j], oc[j]);
            }
            const float qsc = qdec[mi * 16 + (lane & 15)];
#pragma unroll
            for (int kk = 0; kk < 128; kk += 32) {
                const v16bf a = load_a_frag_scaled(q_rm, mi * 16, kk, 128, lane, qsc);
                v16bf bfr[4];
#pragma unroll
                for (int j = 0; j < 4; ++j)
                    bfr[j] = load_b_frag_f32(stateT, (nb + j) * 16, kk, 128, lane);
#pragma unroll
                for (int j = 0; j < 4; ++j)
                    oc[j] = wmma_bf16(a, bfr[j], oc[j]);
            }
#pragma unroll
            for (int j = 0; j < 4; ++j)
#pragma unroll
                for (int r = 0; r < 8; ++r) {
                    const int i_ = mi * 16 + r + ((lane >> 4) << 3);
                    const int d_ = (nb + j) * 16 + (lane & 15);
                    o[(rowbase + i_) * NHD + h * HDIM + d_] = oc[j][r];
                }
        }
        __syncthreads();

        // ---- Phase 3: stateT = lam*stateT + (v^T) @ (k*k_dec) ----
        {
            const int me = wave;
            const v16bf a0 = load_a_frag(vT, me * 16,  0, 64, lane);
            const v16bf a1 = load_a_frag(vT, me * 16, 32, 64, lane);
#pragma unroll
            for (int jt = 0; jt < 8; ++jt) {
                v8f cc;
#pragma unroll
                for (int r = 0; r < 8; ++r) {
                    const int e_ = me * 16 + r + ((lane >> 4) << 3);
                    const int d_ = jt * 16 + (lane & 15);
                    cc[r] = stateT[e_ * 128 + d_] * lam;
                }
                const v16bf b0 = load_b_frag(kdT, jt * 16,  0, 64, lane);
                const v16bf b1 = load_b_frag(kdT, jt * 16, 32, 64, lane);
                cc = wmma_bf16(a0, b0, cc);
                cc = wmma_bf16(a1, b1, cc);
#pragma unroll
                for (int r = 0; r < 8; ++r) {
                    const int e_ = me * 16 + r + ((lane >> 4) << 3);
                    const int d_ = jt * 16 + (lane & 15);
                    stateT[e_ * 128 + d_] = cc[r];
                }
            }
        }
        __syncthreads();
    }
}

// ---------------------------------------------------------------------------
// Group RMSNorm (8 groups of 256) + g_norm_w + sigmoid gate -> bf16
// ---------------------------------------------------------------------------
__global__ __launch_bounds__(256) void gnorm_gate(const float* __restrict__ o,
                                                  const bf16_t* __restrict__ gp,
                                                  const float* __restrict__ gw,
                                                  bf16_t* __restrict__ og) {
    const int g = blockIdx.x & 7;
    const size_t row = (size_t)(blockIdx.x >> 3);
    const int t = threadIdx.x;
    const int col = g * 256 + t;

    const float x = o[row * NHD + col];
    __shared__ float red[8];
    float s = x * x;
#pragma unroll
    for (int off = 16; off; off >>= 1) s += __shfl_xor(s, off);
    if ((t & 31) == 0) red[t >> 5] = s;
    __syncthreads();
    float sum = 0.f;
#pragma unroll
    for (int i = 0; i < 8; ++i) sum += red[i];

    const float r = rsqrtf(sum * (1.0f / 256.0f) + 1e-6f);
    const float gate = 1.0f / (1.0f + expf(-(float)gp[row * NHD + col]));
    og[row * NHD + col] = (bf16_t)(x * r * gw[col] * gate);
}

// ---------------------------------------------------------------------------
// Launch
// ---------------------------------------------------------------------------
extern "C" void kernel_launch(void* const* d_in, const int* in_sizes, int n_in,
                              void* d_out, int out_size, void* d_ws, size_t ws_size,
                              hipStream_t stream) {
    (void)in_sizes; (void)n_in; (void)out_size; (void)ws_size;

    const int*   positions = (const int*)  d_in[0];
    const float* hidden    = (const float*)d_in[1];
    const float* recstate  = (const float*)d_in[2];
    const float* w_qkv     = (const float*)d_in[3];
    const float* w_g       = (const float*)d_in[4];
    const float* w_dense   = (const float*)d_in[5];
    const float* q_norm_w  = (const float*)d_in[6];
    const float* k_norm_w  = (const float*)d_in[7];
    const float* g_norm_w  = (const float*)d_in[8];
    float* out = (float*)d_out;

    const size_t ROWS = (size_t)BATCH * SEQLEN;  // 4096
    char* p = (char*)d_ws;
    auto take = [&](size_t bytes) {
        char* r = p;
        p += (bytes + 255) & ~(size_t)255;
        return r;
    };
    bf16_t* hb    = (bf16_t*)take(ROWS * HIDDEN * 2);
    bf16_t* wqkvb = (bf16_t*)take((size_t)HIDDEN * QKVN * 2);
    bf16_t* wgb   = (bf16_t*)take((size_t)HIDDEN * NHD * 2);
    bf16_t* wdb   = (bf16_t*)take((size_t)NHD * HIDDEN * 2);
    bf16_t* qkvb  = (bf16_t*)take(ROWS * QKVN * 2);
    bf16_t* gpb   = (bf16_t*)take(ROWS * NHD * 2);
    bf16_t* qrb   = (bf16_t*)take(ROWS * NHD * 2);
    bf16_t* krb   = (bf16_t*)take(ROWS * NHD * 2);
    float*  ofb   = (float*) take(ROWS * NHD * 4);
    bf16_t* ogb   = (bf16_t*)take(ROWS * NHD * 2);

    cvt_f32_bf16<<<2048, 256, 0, stream>>>(hidden,  hb,    ROWS * HIDDEN);
    cvt_f32_bf16<<<4096, 256, 0, stream>>>(w_qkv,   wqkvb, (size_t)HIDDEN * QKVN);
    cvt_f32_bf16<<<2048, 256, 0, stream>>>(w_g,     wgb,   (size_t)HIDDEN * NHD);
    cvt_f32_bf16<<<2048, 256, 0, stream>>>(w_dense, wdb,   (size_t)NHD * HIDDEN);

    gemm_bf16<<<dim3(QKVN / 128, ROWS / 128), 256, 0, stream>>>(
        hb, wqkvb, nullptr, qkvb, (int)ROWS, QKVN, HIDDEN);
    gemm_bf16<<<dim3(NHD / 128, ROWS / 128), 256, 0, stream>>>(
        hb, wgb, nullptr, gpb, (int)ROWS, NHD, HIDDEN);

    qk_norm_rope<<<BATCH * SEQLEN * NHEAD, 128, 0, stream>>>(
        qkvb, positions, q_norm_w, k_norm_w, qrb, krb);

    const size_t attn_smem =
        65536 /*stateT*/ + 4 * 16384 /*q,k,kdT,vT*/ + 8192 /*s_bf*/ + 512 /*decays*/;
    attn_kernel<<<BATCH * NHEAD, 256, attn_smem, stream>>>(
        qrb, krb, qkvb, recstate, ofb);

    gnorm_gate<<<BATCH * SEQLEN * 8, 256, 0, stream>>>(ofb, gpb, g_norm_w, ogb);

    gemm_bf16<<<dim3(HIDDEN / 128, ROWS / 128), 256, 0, stream>>>(
        ogb, wdb, out, nullptr, (int)ROWS, HIDDEN, NHD);
}
